// Attention_13297218748956
// MI455X (gfx1250) — matrix operand order repaired and checked by reading, not compiled, not run on hardware
//
#include <hip/hip_runtime.h>

// ---------------------------------------------------------------------------
// Types for WMMA fragments (gfx1250 wave32)
// ---------------------------------------------------------------------------
typedef __bf16 v16bf __attribute__((ext_vector_type(16)));
typedef float  v8f   __attribute__((ext_vector_type(8)));
typedef unsigned short ush8 __attribute__((ext_vector_type(8)));
typedef unsigned int   u32x4 __attribute__((ext_vector_type(4)));
typedef int            i32x4 __attribute__((ext_vector_type(4)));
typedef int            i32x8 __attribute__((ext_vector_type(8)));

union Frag {
    ush8  u8[2];   // two 16-byte loads
    v16bf bf;      // WMMA operand view
};

// float -> bf16 round-to-nearest-even
static __device__ __forceinline__ unsigned short f2bf(float f) {
    unsigned int u = __float_as_uint(f);
    u += 0x7FFFu + ((u >> 16) & 1u);
    return (unsigned short)(u >> 16);
}
static __device__ __forceinline__ float bf2f(unsigned short h) {
    return __uint_as_float(((unsigned int)h) << 16);
}

static __device__ __forceinline__ v8f wmma_bf16(const Frag& a, const Frag& b, v8f c) {
    return __builtin_amdgcn_wmma_f32_16x16x32_bf16(
        /*neg_a=*/false, a.bf, /*neg_b=*/false, b.bf,
        /*c_mod=*/(short)0, c, /*reuse_a=*/false, /*reuse_b=*/false);
}

// ---------------------------------------------------------------------------
// Tensor Data Mover: 2D bf16 tile (tile_d1 rows x tile_d0 elems) from global
// (row stride = stride_elems) into LDS at lds_off, padding +16B every 256B
// so LDS row stride = tile_d0*2 + 16 bytes (136 ushorts for tile_d0=128).
// 6-arg builtin (clang-23 form); descriptor per CDNA5 ISA.
// ---------------------------------------------------------------------------
static __device__ __forceinline__ void tdm_load_2d_bf16(
    unsigned lds_off, const unsigned short* gptr,
    unsigned tile_d0, unsigned tile_d1, unsigned stride_elems)
{
    const unsigned long long ga = (unsigned long long)(uintptr_t)gptr;
    u32x4 g0;
    g0[0] = 1u;                                          // count=1 (valid, user)
    g0[1] = lds_off;                                     // lds_addr [63:32]
    g0[2] = (unsigned)(ga & 0xFFFFFFFFull);              // global_addr [95:64]
    g0[3] = (unsigned)((ga >> 32) & 0x1FFFFFFull)        // global_addr [120:96]
          | (2u << 30);                                  // type=2 ("image")
    i32x8 g1;
    g1[0] = (1 << 16)                                    // data_size = 2 bytes
          | (1 << 20)                                    // pad_enable
          | (5 << 22)                                    // pad_interval: 64 DW (256B)
          | (3 << 25);                                   // pad_amount: 4 DW (16B)
    g1[1] = (int)((tile_d0 & 0xFFFFu) << 16);            // tensor_dim0 [15:0]
    g1[2] = (int)(((tile_d0 >> 16) & 0xFFFFu)            // tensor_dim0 [31:16]
          | ((tile_d1 & 0xFFFFu) << 16));                // tensor_dim1 [15:0]
    g1[3] = (int)(((tile_d1 >> 16) & 0xFFFFu)            // tensor_dim1 [31:16]
          | ((tile_d0 & 0xFFFFu) << 16));                // tile_dim0
    g1[4] = (int)(tile_d1 & 0xFFFFu);                    // tile_dim1 (tile_dim2=0)
    g1[5] = (int)stride_elems;                           // tensor_dim0_stride [31:0]
    g1[6] = 0;
    g1[7] = 0;
    i32x4 gz4 = {0, 0, 0, 0};
    i32x8 gz8 = {0, 0, 0, 0, 0, 0, 0, 0};
    __builtin_amdgcn_tensor_load_to_lds(g0, g1, gz4, gz4, gz8, 0);
}

// ---------------------------------------------------------------------------
// Kernel 1: f32 -> bf16 cast (grid-stride)
// ---------------------------------------------------------------------------
__global__ void cast_f32_bf16(const float* __restrict__ src,
                              unsigned short* __restrict__ dst, size_t n) {
    size_t i      = (size_t)blockIdx.x * blockDim.x + threadIdx.x;
    size_t stride = (size_t)gridDim.x * blockDim.x;
    for (; i < n; i += stride) dst[i] = f2bf(src[i]);
}

// ---------------------------------------------------------------------------
// Kernel 2: GEMM  D[M,N] = A[M,K] * W[N,K]^T   (bf16 in, f32 accum)
// 256 threads = 8 waves; block tile 128(M) x 256(N); wave tile 64x64.
// Both operands are K-contiguous in memory (A rows, W rows).
// ---------------------------------------------------------------------------
template <bool OUT_F32>
__global__ __launch_bounds__(256) void gemm_bf16(
    const unsigned short* __restrict__ A,   // [M,K] bf16
    const unsigned short* __restrict__ W,   // [N,K] bf16
    void* __restrict__ Dout,                // [M,N] f32 or bf16
    int M, int N, int K)
{
    const int tid  = threadIdx.x;
    const int wid  = tid >> 5;
    const int lane = tid & 31;
    const int lm   = lane & 15;   // row/col within 16
    const int lh   = lane >> 4;   // lane half selects K chunk

    const int m0 = blockIdx.y * 128 + (wid & 1) * 64;
    const int n0 = blockIdx.x * 256 + (wid >> 1) * 64;

    v8f acc[4][4] = {};

    for (int k = 0; k < K; k += 32) {
        Frag a[4], b[4];
#pragma unroll
        for (int i = 0; i < 4; i++) {
            const unsigned short* ap = A + (size_t)(m0 + 16 * i + lm) * K + k + 8 * lh;
            a[i].u8[0] = *(const ush8*)(ap);        // K = [8*lh, 8*lh+8)
            a[i].u8[1] = *(const ush8*)(ap + 16);   // K = [16+8*lh, ..+8)
            if (k + 64 < K) __builtin_prefetch(ap + 64, 0, 1);
        }
#pragma unroll
        for (int j = 0; j < 4; j++) {
            const unsigned short* bp = W + (size_t)(n0 + 16 * j + lm) * K + k + 8 * lh;
            b[j].u8[0] = *(const ush8*)(bp);        // K = [8*lh, 8*lh+8)
            b[j].u8[1] = *(const ush8*)(bp + 16);   // K = [16+8*lh, ..+8)
            if (k + 64 < K) __builtin_prefetch(bp + 64, 0, 1);
        }
#pragma unroll
        for (int i = 0; i < 4; i++)
#pragma unroll
            for (int j = 0; j < 4; j++)
                acc[i][j] = wmma_bf16(a[i], b[j], acc[i][j]);
    }

#pragma unroll
    for (int i = 0; i < 4; i++)
#pragma unroll
        for (int j = 0; j < 4; j++)
#pragma unroll
            for (int r = 0; r < 8; r++) {
                const int m = m0 + 16 * i + r + 8 * lh;   // D: vgpr r -> M=r / M=8+r
                const int n = n0 + 16 * j + lm;
                const float v = acc[i][j][r];
                if (OUT_F32)
                    ((float*)Dout)[(size_t)m * N + n] = v;
                else
                    ((unsigned short*)Dout)[(size_t)m * N + n] = f2bf(v);
            }
}

// ---------------------------------------------------------------------------
// Kernel 3: in-place RoPE on bf16 tensor [B*T, C], pairs (2j,2j+1) within head
// One u32 load/store per pair (4B aligned).
// ---------------------------------------------------------------------------
__global__ void rope_kernel(unsigned int* __restrict__ qk,   // pair view
                            const float* __restrict__ cosb,
                            const float* __restrict__ sinb,
                            int total_pairs) {
    int p = blockIdx.x * blockDim.x + threadIdx.x;
    if (p >= total_pairs) return;
    const int m = p >> 10;         // row (b*T + t);  C/2 = 1024 pairs per row
    const int cp = p & 1023;       // pair index within row
    const int t = m & 127;         // t = m % T (T = 128)
    const int j = cp & 63;         // pair index within head (hd/2 = 64)
    const float c = cosb[t * 64 + j];
    const float s = sinb[t * 64 + j];
    const unsigned int v = qk[p];
    const float e = bf2f((unsigned short)(v & 0xFFFFu));
    const float o = bf2f((unsigned short)(v >> 16));
    const unsigned int lo = f2bf(e * c - o * s);
    const unsigned int hi = f2bf(e * s + o * c);
    qk[p] = lo | (hi << 16);
}

// ---------------------------------------------------------------------------
// Kernel 4: attention per (b,h).  T=128, hd=128.
//   scores = Q K^T / sqrt(hd) (causal) -> softmax -> P V
// LDS: padded f32 score tile (stride 132 f32 = 528B, 16B-aligned rows)
//      transposed V tile     (stride 136 u16 = 272B, 16B-aligned rows)
//      K tile via TDM        (stride 136 u16 via descriptor padding)
// ---------------------------------------------------------------------------
#define SC_STRIDE 132
#define VT_STRIDE 136

__global__ __launch_bounds__(256) void attn_kernel(
    const unsigned short* __restrict__ qb,
    const unsigned short* __restrict__ kb,
    const unsigned short* __restrict__ vb,
    unsigned short* __restrict__ ab)
{
    __shared__ float          sc[128 * SC_STRIDE];     // 67,584 B (P bf16 overlaid)
    __shared__ unsigned short Vt[128 * VT_STRIDE];     // 34,816 B
    __shared__ unsigned short Klds[128 * VT_STRIDE];   // 34,816 B (TDM target)

    const int T = 128, C = 2048, HD = 128;
    const int bh = blockIdx.x;
    const int b  = bh >> 4;
    const int h  = bh & 15;

    const int tid  = threadIdx.x;
    const int wid  = tid >> 5;
    const int lane = tid & 31;
    const int lm   = lane & 15;
    const int lh   = lane >> 4;
    const int t0   = wid * 16;                 // 16-row strip per wave

    const size_t base = (size_t)b * T * C + (size_t)h * HD;

    // Stage 0a: async DMA of K tile (128x128 bf16, row stride C) into LDS,
    // padded to 272B rows (VT_STRIDE). Issued by wave 0 only.
    if (wid == 0) {
        tdm_load_2d_bf16((unsigned)(uintptr_t)&Klds[0], kb + base,
                         /*tile_d0=*/HD, /*tile_d1=*/T, /*stride=*/C);
    }

    // Stage 0b: V -> LDS transposed  Vt[d][s]  (overlaps with TDM)
    for (int idx = tid; idx < T * HD; idx += 256) {
        const int s = idx >> 7;
        const int d = idx & 127;
        Vt[d * VT_STRIDE + s] = vb[base + (size_t)s * C + d];
    }

    if (wid == 0) __builtin_amdgcn_s_wait_tensorcnt(0);
    __syncthreads();

    // Stage 1: scores = Q K^T  (wave: rows [t0,t0+16), all 8 column tiles)
    {
        v8f sacc[8] = {};
        for (int k = 0; k < HD; k += 32) {
            Frag a;
            const unsigned short* ap = qb + base + (size_t)(t0 + lm) * C + k + 8 * lh;
            a.u8[0] = *(const ush8*)(ap);
            a.u8[1] = *(const ush8*)(ap + 16);
#pragma unroll
            for (int j = 0; j < 8; j++) {
                Frag bfr;
                const unsigned short* bp = Klds + (16 * j + lm) * VT_STRIDE + k + 8 * lh;
                bfr.u8[0] = *(const ush8*)(bp);
                bfr.u8[1] = *(const ush8*)(bp + 16);
                sacc[j] = wmma_bf16(a, bfr, sacc[j]);
            }
        }
        const float scale = 0.08838834764831845f;   // 1/sqrt(128)
#pragma unroll
        for (int j = 0; j < 8; j++)
#pragma unroll
            for (int r = 0; r < 8; r++) {
                const int t = t0 + r + 8 * lh;
                const int s = 16 * j + lm;
                float v = sacc[j][r] * scale;
                if (s > t) v = -1e30f;               // causal mask
                sc[t * SC_STRIDE + s] = v;
            }
    }
    __syncthreads();

    // Stage 2: per-row softmax; write P (bf16) overlaid on the f32 row
    if (tid < T) {
        float* row = sc + tid * SC_STRIDE;
        float mx = -1e30f;
        for (int s = 0; s < T; s++) mx = fmaxf(mx, row[s]);
        float sum = 0.0f;
        for (int s = 0; s < T; s++) sum += __expf(row[s] - mx);
        const float inv = 1.0f / sum;
        unsigned short* prow = (unsigned short*)row;
        // prow[s] clobbers only f32 element s/2 (< s): safe read-then-write order
        for (int s = 0; s < T; s++) prow[s] = f2bf(__expf(row[s] - mx) * inv);
    }
    __syncthreads();

    // Stage 3: out = P V  (A rows from LDS P, B from transposed V in LDS)
    {
        v8f oacc[8] = {};
        for (int k = 0; k < T; k += 32) {
            Frag a;
            const unsigned short* ap =
                (const unsigned short*)(sc + (t0 + lm) * SC_STRIDE) + k + 8 * lh;
            a.u8[0] = *(const ush8*)(ap);
            a.u8[1] = *(const ush8*)(ap + 16);
#pragma unroll
            for (int j = 0; j < 8; j++) {
                Frag bfr;
                const unsigned short* bp = Vt + (16 * j + lm) * VT_STRIDE + k + 8 * lh;
                bfr.u8[0] = *(const ush8*)(bp);
                bfr.u8[1] = *(const ush8*)(bp + 16);
                oacc[j] = wmma_bf16(a, bfr, oacc[j]);
            }
        }
#pragma unroll
        for (int j = 0; j < 8; j++)
#pragma unroll
            for (int r = 0; r < 8; r++) {
                const int t = t0 + r + 8 * lh;
                const int d = 16 * j + lm;
                ab[base + (size_t)t * C + d] = f2bf(oacc[j][r]);
            }
    }
}

// ---------------------------------------------------------------------------
// Host-side launcher
// ---------------------------------------------------------------------------
extern "C" void kernel_launch(void* const* d_in, const int* in_sizes, int n_in,
                              void* d_out, int out_size, void* d_ws, size_t ws_size,
                              hipStream_t stream) {
    const int B = 64, T = 128, C = 2048, H = 16;
    const size_t M  = (size_t)B * T;          // 8192
    const size_t MC = M * C;                  // 16,777,216
    const size_t CC = (size_t)C * C;          // 4,194,304

    const float* x  = (const float*)d_in[0];
    const float* fc = (const float*)d_in[1];
    const float* fs = (const float*)d_in[2];
    const float* wq = (const float*)d_in[3];
    const float* wk = (const float*)d_in[4];
    const float* wv = (const float*)d_in[5];
    const float* wo = (const float*)d_in[6];
    float* out = (float*)d_out;

    // Workspace layout (bf16 buffers), ~192 MB total
    char* ws = (char*)d_ws;
    size_t off = 0;
    unsigned short* xb  = (unsigned short*)(ws + off); off += MC * 2;
    unsigned short* wqb = (unsigned short*)(ws + off); off += CC * 2;
    unsigned short* wkb = (unsigned short*)(ws + off); off += CC * 2;
    unsigned short* wvb = (unsigned short*)(ws + off); off += CC * 2;
    unsigned short* wob = (unsigned short*)(ws + off); off += CC * 2;
    unsigned short* qb  = (unsigned short*)(ws + off); off += MC * 2;
    unsigned short* kb  = (unsigned short*)(ws + off); off += MC * 2;
    unsigned short* vb  = (unsigned short*)(ws + off); off += MC * 2;
    unsigned short* ab  = (unsigned short*)(ws + off); off += MC * 2;

    // 1) casts to bf16
    cast_f32_bf16<<<4096, 256, 0, stream>>>(x,  xb,  MC);
    cast_f32_bf16<<<2048, 256, 0, stream>>>(wq, wqb, CC);
    cast_f32_bf16<<<2048, 256, 0, stream>>>(wk, wkb, CC);
    cast_f32_bf16<<<2048, 256, 0, stream>>>(wv, wvb, CC);
    cast_f32_bf16<<<2048, 256, 0, stream>>>(wo, wob, CC);

    // 2) Q/K/V projections: [8192,2048] = xb[8192,2048] @ w[2048,2048]^T
    dim3 gg(C / 256, (int)(M / 128));         // (8, 64)
    gemm_bf16<false><<<gg, 256, 0, stream>>>(xb, wqb, qb, (int)M, C, C);
    gemm_bf16<false><<<gg, 256, 0, stream>>>(xb, wkb, kb, (int)M, C, C);
    gemm_bf16<false><<<gg, 256, 0, stream>>>(xb, wvb, vb, (int)M, C, C);

    // 3) RoPE on Q and K (in place)
    const int pairs = (int)(MC / 2);          // 8,388,608
    rope_kernel<<<(pairs + 255) / 256, 256, 0, stream>>>((unsigned int*)qb, fc, fs, pairs);
    rope_kernel<<<(pairs + 255) / 256, 256, 0, stream>>>((unsigned int*)kb, fc, fs, pairs);

    // 4) attention: one workgroup per (b,h)
    attn_kernel<<<B * H, 256, 0, stream>>>(qb, kb, vb, ab);

    // 5) output projection -> f32 d_out
    gemm_bf16<true><<<gg, 256, 0, stream>>>(ab, wob, out, (int)M, C, C);
}
